// Outer_52570399703542
// MI455X (gfx1250) — matrix operand order repaired and checked by reading, not compiled, not run on hardware
//
#include <hip/hip_runtime.h>
#include <stdint.h>

// ---------------------------------------------------------------------------
// MI455X / gfx1250 implementation.
//
//   out = relu(relu((a (x) b) @ W1 + b1) @ W2 + b2)
//
// Never materialize the 1.08 GB fusion tensor. Repack W1 (fp32, 67.6 MB) into
// bf16 "W1p" with the j-dimension padded 257->288 and K-pairs interleaved:
// W1p[(i*288+j)/2][n][2]. With zero-padded extended input vectors, layer 1 is
// one regular GEMM, K'=74048 (2314 chunks of 32, incl. 1 zero pad chunk),
// computed with v_wmma_f32_16x16x32_bf16. A-fragments are generated on the
// fly with v_pk_mul_bf16: A[b,k'] = a_ext[b, k'/288] * b_ext[b, k'%288],
// where a_ext is pre-duplicated packed bf16 (one b32 load per chunk).
// W1p (~36 MB bf16) stays L2-resident; 8 KB tiles (2 K-chunks) stream to LDS
// via global_load_async_to_lds_b128, triple buffered (2 tiles in flight) on
// s_wait_asynccnt with a single workgroup barrier per tile.
// ---------------------------------------------------------------------------

typedef __bf16 v16bf  __attribute__((ext_vector_type(16)));
typedef __bf16 bf16x2 __attribute__((ext_vector_type(2)));
typedef float  v8f    __attribute__((ext_vector_type(8)));
typedef int    v4i    __attribute__((ext_vector_type(4)));

#define K_PAD     288                  // padded j range (multiple of 32)
#define NCHUNK    2314                 // ceil(257*288/32) padded to even
#define NPAIR_W1  (NCHUNK * 16)        // 37024 K-pairs (incl. zero pad)
#define NTILES    (NCHUNK / 2)         // 1157 staged tiles (32 pairs each)
#define A_STRIDE  288                  // a_ext row stride (dwords)

union Frag {
  unsigned int u[8];
  v16bf v;
};

// ---- bf16 helpers (round to nearest even) ---------------------------------
__device__ __forceinline__ unsigned int f2bf(float f) {
  unsigned int u = __builtin_bit_cast(unsigned int, f);
  u += 0x7fffu + ((u >> 16) & 1u);
  return u >> 16;                       // 16-bit result in low bits
}

// packed bf16 pair * broadcast bf16 pair  ->  v_pk_mul_bf16
__device__ __forceinline__ unsigned int pkmul2(bf16x2 a2, unsigned int p) {
  bf16x2 b2 = __builtin_bit_cast(bf16x2, p);
  bf16x2 r = a2 * b2;
  return __builtin_bit_cast(unsigned int, r);
}

// ---- async global->LDS b128 copy (fallback: sync copy) --------------------
#if __has_builtin(__builtin_amdgcn_global_load_async_to_lds_b128)
#define HAVE_ASYNC_LDS 1
#else
#define HAVE_ASYNC_LDS 0
#endif

__device__ __forceinline__ void cp_b128_to_lds(const unsigned int* g,
                                               unsigned int* l) {
#if HAVE_ASYNC_LDS
  __builtin_amdgcn_global_load_async_to_lds_b128(
      (__attribute__((address_space(1))) v4i*)(unsigned int*)g,
      (__attribute__((address_space(3))) v4i*)l, 0, 0);
#else
  *(uint4*)l = *(const uint4*)g;
#endif
}

__device__ __forceinline__ void wait_async_le2() {
#if HAVE_ASYNC_LDS
#if __has_builtin(__builtin_amdgcn_s_wait_asynccnt)
  __builtin_amdgcn_s_wait_asynccnt(2);
#else
  asm volatile("s_wait_asynccnt 0x2" ::: "memory");
#endif
#endif
}
__device__ __forceinline__ void wait_async_le0() {
#if HAVE_ASYNC_LDS
#if __has_builtin(__builtin_amdgcn_s_wait_asynccnt)
  __builtin_amdgcn_s_wait_asynccnt(0);
#else
  asm volatile("s_wait_asynccnt 0x0" ::: "memory");
#endif
#endif
}

// ---------------------------------------------------------------------------
// Prep: a_ext (pre-duplicated packed bf16): adup[row][p] =
//   p<256: {bf16(inp1[row][p]), bf16(inp1[row][p])}; p==256: {1,1}; else 0.
// ---------------------------------------------------------------------------
__global__ void __launch_bounds__(256) prep_a_kernel(
    const float* __restrict__ inp1, unsigned int* __restrict__ adup) {
  const int total = 4096 * A_STRIDE;
  for (int idx = blockIdx.x * 256 + threadIdx.x; idx < total;
       idx += gridDim.x * 256) {
    const int row = idx / A_STRIDE;
    const int p = idx - row * A_STRIDE;
    unsigned int v = 0u;
    if (p < 256) {
      const unsigned int b = f2bf(inp1[row * 256 + p]);
      v = b | (b << 16);
    } else if (p == 256) {
      v = 0x3f803f80u;
    }
    adup[idx] = v;
  }
}

// ---------------------------------------------------------------------------
// Prep: W1 (fp32 [66049,256]) -> W1p (bf16 pairs, [NPAIR_W1][256] dwords)
//   k' = 2*pairIdx (+e);  i = k'/288, j = k'%288;  src row = i*257+j; pad=0.
// ---------------------------------------------------------------------------
__global__ void __launch_bounds__(256) prep_w1_kernel(
    const float* __restrict__ W1, unsigned int* __restrict__ W1p) {
  const int total = NPAIR_W1 * 256;
  for (int idx = blockIdx.x * 256 + threadIdx.x; idx < total;
       idx += gridDim.x * 256) {
    const int pairIdx = idx >> 8;
    const int n = idx & 255;
    const int kp = pairIdx * 2;
    const int i = kp / K_PAD;
    const int j = kp - i * K_PAD;
    float lo = 0.f, hi = 0.f;
    if (i <= 256 && j <= 256) lo = W1[(i * 257 + j) * 256 + n];
    if (i <= 256 && j + 1 <= 256) hi = W1[(i * 257 + j + 1) * 256 + n];
    W1p[idx] = (f2bf(lo) & 0xffffu) | (f2bf(hi) << 16);
  }
}

// W2 (fp32 [256,256]) -> W2p (bf16 pairs, [128][256] dwords)
__global__ void __launch_bounds__(256) prep_w2_kernel(
    const float* __restrict__ W2, unsigned int* __restrict__ W2p) {
  const int idx = blockIdx.x * 256 + threadIdx.x;   // 128*256 total
  const int pair = idx >> 8;
  const int n = idx & 255;
  const float lo = W2[(2 * pair) * 256 + n];
  const float hi = W2[(2 * pair + 1) * 256 + n];
  W2p[idx] = (f2bf(lo) & 0xffffu) | (f2bf(hi) << 16);
}

// ---------------------------------------------------------------------------
// Layer 1:  h = relu(fusion @ W1 + b1)   (h stored bf16 [4096,256])
// Grid: (B/64, N/64) = (64,4).  256 threads = 8 waves; wave (wb,wn) owns a
// 16x32 output tile (2 WMMA tiles, 16 accum VGPRs).
// ---------------------------------------------------------------------------
__global__ void __launch_bounds__(256) layer1_kernel(
    const unsigned int* __restrict__ adup, const float* __restrict__ inp2,
    const unsigned int* __restrict__ W1p, const float* __restrict__ b1,
    unsigned short* __restrict__ hout) {
  __shared__ unsigned int beP[64 * 144];       // b_ext bf16 pairs, 36 KB
  __shared__ unsigned int btile[3][32 * 64];   // W1p tiles, 3 x 8 KB

  const int tid  = threadIdx.x;
  const int lane = tid & 31;
  const int wave = tid >> 5;
  const int wb   = wave & 3;        // batch sub-tile (x16 rows)
  const int wn   = wave >> 2;       // N sub-tile (x32 cols)
  const int half = lane >> 4;
  const int M    = lane & 15;
  const int row0 = blockIdx.x * 64;
  const int n0   = blockIdx.y * 64;

  // Stage extended/padded inp2 rows as packed bf16 pairs.
  for (int idx = tid; idx < 64 * 144; idx += 256) {
    const int r = idx / 144;
    const int p = idx - r * 144;
    unsigned int val = 0u;
    if (p < 128) {
      const float* src = inp2 + (row0 + r) * 256 + 2 * p;
      val = (f2bf(src[0]) & 0xffffu) | (f2bf(src[1]) << 16);
    } else if (p == 128) {
      val = 0x3f80u;                  // (1.0, 0.0) : the ones column
    }
    beP[idx] = val;
  }

  const unsigned int* arow = adup + (row0 + wb * 16 + M) * A_STRIDE;
  const int srow = tid >> 3;          // 32 K-pairs per tile
  const int schk = tid & 7;           // 8 x 16B chunks per half row
  const int rowbase = (wb * 16 + M) * 144 + half * 4;
  const int bc = wn * 32 + M;

  v8f acc0 = {}, acc1 = {};

  // prologue: stage tiles 0 and 1 (2 async b128 per thread per tile)
#pragma unroll
  for (int t = 0; t < 2; ++t) {
    const unsigned int* g = W1p + (t * 32 + srow) * 256 + n0 + schk * 4;
    unsigned int* l = &btile[t][srow * 64 + schk * 4];
    cp_b128_to_lds(g, l);
    cp_b128_to_lds(g + 32, l + 32);
  }
  __syncthreads();                    // covers beP too

  int ii = 0, jc = 0, buf = 0, nbuf = 2;
  for (int t = 0; t < NTILES; ++t) {
    if (t + 1 < NTILES) wait_async_le2();  // tile t landed; t+1 in flight
    else                wait_async_le0();
    __syncthreads();                  // all waves' tile t visible; also
                                      // orders compute(t-1) before stage(t+2)

    const unsigned int* bt = btile[buf];
#pragma unroll
    for (int c2 = 0; c2 < 2; ++c2) {
      // broadcast a_ext[row, ii] (already packed bf16 pair)
      const bf16x2 a2 = __builtin_bit_cast(bf16x2, arow[ii]);

      // A fragment: 16-bit A 16x32 layout; pair index per VGPR v:
      //   p = (v<4?0:8) + half*4 + (v&3)
      Frag A, B0, B1;
      const int bebase = rowbase + jc * 16;
#pragma unroll
      for (int v = 0; v < 8; ++v) {
        const int p = ((v & 4) << 1) + (v & 3);
        A.u[v] = pkmul2(a2, beP[bebase + p]);
      }

      // B fragment: VGPR v holds K-pair ((v<4?0:8) + half*4 + (v&3)) at column N=lane%16.
      const unsigned int* btc = bt + c2 * (16 * 64);
#pragma unroll
      for (int v = 0; v < 8; ++v) {
        const int pl = (((v & 4) << 1) + half * 4 + (v & 3)) * 64;
        B0.u[v] = btc[pl + bc];
        B1.u[v] = btc[pl + bc + 16];
      }

      acc0 = __builtin_amdgcn_wmma_f32_16x16x32_bf16(
          false, A.v, false, B0.v, (short)0, acc0, false, false);
      acc1 = __builtin_amdgcn_wmma_f32_16x16x32_bf16(
          false, A.v, false, B1.v, (short)0, acc1, false, false);

      if (++jc == 9) { jc = 0; ++ii; }
    }

    // stage tile t+2 into the buffer compute(t-1) just vacated; the barrier
    // above ordered every thread's compute(t-1) before this write.
    if (t + 2 < NTILES) {
      const unsigned int* g = W1p + ((t + 2) * 32 + srow) * 256 + n0 + schk * 4;
      unsigned int* l = &btile[nbuf][srow * 64 + schk * 4];
      cp_b128_to_lds(g, l);
      cp_b128_to_lds(g + 32, l + 32);
    }
    buf = (buf == 2) ? 0 : buf + 1;
    nbuf = (nbuf == 2) ? 0 : nbuf + 1;
  }

  // Epilogue: + b1, ReLU, store bf16. C/D layout: N=lane%16, M=r+8*half.
#pragma unroll
  for (int t = 0; t < 2; ++t) {
    const int ng = n0 + wn * 32 + t * 16 + M;
    const float bias = b1[ng];
    const v8f acc = t ? acc1 : acc0;
#pragma unroll
    for (int r = 0; r < 8; ++r) {
      const int gm = row0 + wb * 16 + half * 8 + r;
      float x = acc[r] + bias;
      x = x > 0.f ? x : 0.f;
      hout[gm * 256 + ng] = (unsigned short)f2bf(x);
    }
  }
}

// ---------------------------------------------------------------------------
// Layer 2:  out = relu(h @ W2 + b2)  (fp32 out).  K=256 -> 8 chunks.
// Whole W2p N-slice (32 KB) preloaded to LDS; A loaded straight from global h.
// ---------------------------------------------------------------------------
__global__ void __launch_bounds__(256) layer2_kernel(
    const unsigned int* __restrict__ h32,      // bf16 pairs [4096][128]
    const unsigned int* __restrict__ W2p, const float* __restrict__ b2,
    float* __restrict__ out) {
  __shared__ unsigned int w2l[128 * 64];       // 32 KB

  const int tid  = threadIdx.x;
  const int lane = tid & 31;
  const int wave = tid >> 5;
  const int wb   = wave & 3;
  const int wn   = wave >> 2;
  const int half = lane >> 4;
  const int M    = lane & 15;
  const int row0 = blockIdx.x * 64;
  const int n0   = blockIdx.y * 64;

  for (int idx = tid; idx < 128 * 64; idx += 256) {
    const int pair = idx >> 6;
    const int nl = idx & 63;
    w2l[idx] = W2p[pair * 256 + n0 + nl];
  }
  __syncthreads();

  const unsigned int* hrow = h32 + (row0 + wb * 16 + M) * 128;

  v8f acc0 = {}, acc1 = {};
#pragma unroll
  for (int jc = 0; jc < 8; ++jc) {
    Frag A, B0, B1;
    const int abase = jc * 16 + half * 4;
#pragma unroll
    for (int v = 0; v < 8; ++v)
      A.u[v] = hrow[abase + ((v & 4) << 1) + (v & 3)];
    const int bc = wn * 32 + M;
#pragma unroll
    for (int v = 0; v < 8; ++v) {
      const int pl = (jc * 16 + ((v & 4) << 1) + half * 4 + (v & 3)) * 64;
      B0.u[v] = w2l[pl + bc];
      B1.u[v] = w2l[pl + bc + 16];
    }
    acc0 = __builtin_amdgcn_wmma_f32_16x16x32_bf16(
        false, A.v, false, B0.v, (short)0, acc0, false, false);
    acc1 = __builtin_amdgcn_wmma_f32_16x16x32_bf16(
        false, A.v, false, B1.v, (short)0, acc1, false, false);
  }

#pragma unroll
  for (int t = 0; t < 2; ++t) {
    const int ng = n0 + wn * 32 + t * 16 + M;
    const float bias = b2[ng];
    const v8f acc = t ? acc1 : acc0;
#pragma unroll
    for (int r = 0; r < 8; ++r) {
      const int gm = row0 + wb * 16 + half * 8 + r;
      float x = acc[r] + bias;
      out[gm * 256 + ng] = x > 0.f ? x : 0.f;
    }
  }
}

// ---------------------------------------------------------------------------
extern "C" void kernel_launch(void* const* d_in, const int* in_sizes, int n_in,
                              void* d_out, int out_size, void* d_ws,
                              size_t ws_size, hipStream_t stream) {
  const float* inp1 = (const float*)d_in[0];
  const float* inp2 = (const float*)d_in[1];
  const float* W1   = (const float*)d_in[2];
  const float* b1   = (const float*)d_in[3];
  const float* W2   = (const float*)d_in[4];
  const float* b2   = (const float*)d_in[5];

  // Workspace layout (dwords): W1p | W2p | h(bf16 pairs) | a_ext.  ~42.8 MB.
  unsigned int* W1p  = (unsigned int*)d_ws;
  unsigned int* W2p  = W1p + (size_t)NPAIR_W1 * 256;
  unsigned int* h32  = W2p + 128 * 256;
  unsigned int* adup = h32 + 4096 * 128;

  prep_a_kernel<<<2048, 256, 0, stream>>>(inp1, adup);
  prep_w1_kernel<<<4096, 256, 0, stream>>>(W1, W1p);
  prep_w2_kernel<<<128, 256, 0, stream>>>(W2, W2p);
  layer1_kernel<<<dim3(64, 4), 256, 0, stream>>>(adup, inp2, W1p, b1,
                                                 (unsigned short*)h32);
  layer2_kernel<<<dim3(64, 4), 256, 0, stream>>>(h32, W2p, b2, (float*)d_out);
}
